// GAT_JK_Model_74337293959434
// MI455X (gfx1250) — matrix operand order repaired and checked by reading, not compiled, not run on hardware
//
#include <hip/hip_runtime.h>
#include <cstdint>
#include <cstddef>

typedef __attribute__((ext_vector_type(16))) _Float16 v16h;
typedef __attribute__((ext_vector_type(8)))  float    v8f;

#define GAT_N   50000
#define GAT_E   1600000
#define GAT_ET  (GAT_E + GAT_N)
#define GAT_FIN 128
#define GAT_H   64
#define GAT_OUT 40
#define NEG_SLOPE 0.2f

// ---------------------------------------------------------------------------
// WMMA GEMM: C[M x ncolsStore] = A[M x K] * B[K x ncolsB] (+bias)
// One wave32 per 16-row strip; the wave computes NT 16x16 output tiles
// (NT=4 -> 64 cols, NT=3 -> 48 padded cols), sharing one A fragment across
// NT back-to-back v_wmma_f32_16x16x32_f16 issues per 32-wide K step.
// All tile staging is float4 (global_load_b128 / ds_store_b128).
// ---------------------------------------------------------------------------
template <int NT>
__global__ void wmma_gemm_kernel(const float* __restrict__ A,
                                 const float* __restrict__ B,
                                 float* __restrict__ C,
                                 int K, int ncolsB, int ncolsStore,
                                 const float* __restrict__ bias) {
  __shared__ float4 As4[16 * 8];           // 16 rows x 32 K-cols
  __shared__ float4 Bs4[32 * NT * 4];      // 32 K-rows x NT*16 cols
  const float* As = (const float*)As4;
  const float* Bs = (const float*)Bs4;
  const int lane  = threadIdx.x & 31;
  const int mtile = blockIdx.x;
  const int K4  = K >> 2;
  const int BW  = NT * 16;                 // B tile row width (floats)
  const int BW4 = NT * 4;                  // B tile row width (float4)

  v8f acc[NT];
  #pragma unroll
  for (int t = 0; t < NT; ++t) acc[t] = (v8f){};

  for (int k0 = 0; k0 < K; k0 += 32) {
    // Stage A 16x32 as float4 (8 float4 per row, 4 per lane).
    const float4* A4 = (const float4*)(A + (size_t)mtile * 16 * K + k0);
    #pragma unroll
    for (int i = 0; i < 4; ++i) {
      int idx4 = lane + 32 * i;            // 0..127
      int r = idx4 >> 3, c4 = idx4 & 7;
      As4[idx4] = A4[(size_t)r * K4 + c4];
    }
    // Stage B 32xBW as float4; cols >= ncolsB zero-filled (ncolsB % 4 == 0).
    #pragma unroll
    for (int i = 0; i < BW4; ++i) {
      int idx4 = lane + 32 * i;            // 0 .. 32*BW4-1
      int r = idx4 / BW4, c4 = idx4 - r * BW4;
      int col = c4 * 4;
      float4 v = make_float4(0.f, 0.f, 0.f, 0.f);
      if (col < ncolsB)
        v = *(const float4*)(B + (size_t)(k0 + r) * ncolsB + col);
      Bs4[idx4] = v;
    }
    __syncthreads();

    // A fragment per CDNA5 wave32 layout:
    // lanes 0-15: K akh..akh+7 and 16+akh..; lanes 16-31: +8.
    v16h a;
    const int arow = lane & 15;
    const int akh  = (lane >> 4) * 8;
    #pragma unroll
    for (int h = 0; h < 8; ++h) {
      a[h]     = (_Float16)As[arow * 32 + akh + h];
      a[8 + h] = (_Float16)As[arow * 32 + 16 + akh + h];
    }
    const int bcolbase = lane & 15;
    const int bkb      = (lane >> 4) * 8;   // lanes 0-15: K 0-7,16-23 ; 16-31: K 8-15,24-31
    #pragma unroll
    for (int t = 0; t < NT; ++t) {
      v16h b;
      const int bcol = t * 16 + bcolbase;
      #pragma unroll
      for (int h = 0; h < 16; ++h)
        b[h] = (_Float16)Bs[(h < 8 ? bkb + h : 16 + bkb + (h - 8)) * BW + bcol];
      acc[t] = __builtin_amdgcn_wmma_f32_16x16x32_f16(
          /*neg_a=*/false, a, /*neg_b=*/false, b,
          /*c_mod=*/(short)0, acc[t], /*reuse_a=*/false, /*reuse_b=*/false);
    }
    __syncthreads();
  }

  // C layout: lanes 0-15 -> M = v, N = lane ; lanes 16-31 -> M = 8+v, N = lane-16
  const int mbase = mtile * 16 + ((lane >> 4) * 8);
  #pragma unroll
  for (int t = 0; t < NT; ++t) {
    const int n = t * 16 + (lane & 15);
    if (n < ncolsStore) {
      const float bv = bias ? bias[n] : 0.0f;
      #pragma unroll
      for (int v = 0; v < 8; ++v)
        C[(size_t)(mbase + v) * ncolsStore + n] = acc[t][v] + bv;
    }
  }
}

// ---------------------------------------------------------------------------
// Per-node prep: alpha_s = h·a_src, alpha_d = h·a_dst (wave32 per node, shuffle
// reduction), zero the accumulator row, init segment-max (-inf) and denom.
// ---------------------------------------------------------------------------
__device__ __forceinline__ unsigned fenc(float f) {
  unsigned u = __float_as_uint(f);
  return (u & 0x80000000u) ? ~u : (u | 0x80000000u);   // order-preserving map
}
__device__ __forceinline__ float fdec(unsigned u) {
  return (u & 0x80000000u) ? __uint_as_float(u & 0x7FFFFFFFu) : __uint_as_float(~u);
}

__global__ void node_prep_kernel(const float* __restrict__ h,
                                 const float* __restrict__ asrc,
                                 const float* __restrict__ adst,
                                 float* __restrict__ alpha_s,
                                 float* __restrict__ alpha_d,
                                 unsigned* __restrict__ mmax,
                                 float* __restrict__ denom,
                                 float* __restrict__ accum) {
  const int wid  = (blockIdx.x * blockDim.x + threadIdx.x) >> 5;
  const int lane = threadIdx.x & 31;
  if (wid >= GAT_N) return;
  const float* hr = h + (size_t)wid * GAT_H;
  const float h0 = hr[lane], h1 = hr[lane + 32];
  float s = h0 * asrc[lane] + h1 * asrc[lane + 32];
  float d = h0 * adst[lane] + h1 * adst[lane + 32];
  #pragma unroll
  for (int off = 16; off > 0; off >>= 1) {
    s += __shfl_xor(s, off, 32);
    d += __shfl_xor(d, off, 32);
  }
  accum[(size_t)wid * GAT_H + lane]      = 0.0f;
  accum[(size_t)wid * GAT_H + lane + 32] = 0.0f;
  if (lane == 0) {
    alpha_s[wid] = s;
    alpha_d[wid] = d;
    mmax[wid]    = 0x007FFFFFu;   // fenc(-inf)
    denom[wid]   = 0.0f;
  }
}

// ---------------------------------------------------------------------------
// Edge passes (self-loops appended after the E directed edges).
// ---------------------------------------------------------------------------
__device__ __forceinline__ void edge_sd(int e, const int* __restrict__ ei, int& s, int& d) {
  if (e < GAT_E) { s = ei[e]; d = ei[GAT_E + e]; }
  else           { s = e - GAT_E; d = s; }
}

__global__ void edge_max_kernel(const int* __restrict__ ei,
                                const float* __restrict__ alpha_s,
                                const float* __restrict__ alpha_d,
                                unsigned* __restrict__ mmax) {
  const int e = blockIdx.x * blockDim.x + threadIdx.x;
  if (e >= GAT_ET) return;
  int s, d; edge_sd(e, ei, s, d);
  float v = alpha_s[s] + alpha_d[d];
  v = (v > 0.0f) ? v : NEG_SLOPE * v;
  atomicMax(&mmax[d], fenc(v));
}

__global__ void edge_exp_kernel(const int* __restrict__ ei,
                                const float* __restrict__ alpha_s,
                                const float* __restrict__ alpha_d,
                                const unsigned* __restrict__ mmax,
                                float* __restrict__ p,
                                float* __restrict__ denom) {
  const int e = blockIdx.x * blockDim.x + threadIdx.x;
  if (e >= GAT_ET) return;
  int s, d; edge_sd(e, ei, s, d);
  float v = alpha_s[s] + alpha_d[d];
  v = (v > 0.0f) ? v : NEG_SLOPE * v;
  const float pe = expf(v - fdec(mmax[d]));
  p[e] = pe;
  atomicAdd(&denom[d], pe);
}

// One wave32 per edge: lanes cover the 64-wide feature row as float2.
// h (12.8 MB) and accum stay L2-resident (192 MB L2) -> L2 gathers + L2 atomics.
__global__ void edge_agg_kernel(const int* __restrict__ ei,
                                const float* __restrict__ p,
                                const float* __restrict__ denom,
                                const float* __restrict__ h,
                                float* __restrict__ accum) {
  const int wid  = (blockIdx.x * blockDim.x + threadIdx.x) >> 5;
  const int lane = threadIdx.x & 31;
  if (wid >= GAT_ET) return;
  int s, d; edge_sd(wid, ei, s, d);
  const float coef = p[wid] / denom[d];
  const float2 v = ((const float2*)(h + (size_t)s * GAT_H))[lane];
  float* ar = accum + (size_t)d * GAT_H + lane * 2;
  atomicAdd(ar,     v.x * coef);
  atomicAdd(ar + 1, v.y * coef);
}

// out = relu(accum + bias); xnext = out; jk = first ? out : max(jk, out)
__global__ void act_jk_kernel(const float* __restrict__ accum,
                              const float* __restrict__ bias,
                              float* __restrict__ xnext,
                              float* __restrict__ jk,
                              int first) {
  const int idx = blockIdx.x * blockDim.x + threadIdx.x;
  if (idx >= GAT_N * GAT_H) return;
  const int f = idx & (GAT_H - 1);
  float v = accum[idx] + bias[f];
  v = fmaxf(v, 0.0f);
  xnext[idx] = v;
  jk[idx] = first ? v : fmaxf(jk[idx], v);
}

// ---------------------------------------------------------------------------
extern "C" void kernel_launch(void* const* d_in, const int* in_sizes, int n_in,
                              void* d_out, int out_size, void* d_ws, size_t ws_size,
                              hipStream_t stream) {
  (void)in_sizes; (void)n_in; (void)out_size; (void)ws_size;
  const float* x     = (const float*)d_in[0];   // [N, 128]
  const int*   ei    = (const int*)  d_in[1];   // [2, E]
  const float* W1    = (const float*)d_in[2];   // [128, 64]
  const float* W23   = (const float*)d_in[3];   // [2, 64, 64]
  const float* a_src = (const float*)d_in[4];   // [3, 64]
  const float* a_dst = (const float*)d_in[5];   // [3, 64]
  const float* b     = (const float*)d_in[6];   // [3, 64]
  const float* Wout  = (const float*)d_in[7];   // [64, 40]
  const float* bout  = (const float*)d_in[8];   // [40]
  float* out = (float*)d_out;                   // [N, 40]

  const size_t NH = (size_t)GAT_N * GAT_H;
  float*    ws      = (float*)d_ws;
  float*    hproj   = ws;                       // [N, 64]
  float*    xin     = ws + NH;                  // [N, 64] next-layer input
  float*    accum   = ws + 2 * NH;              // [N, 64]
  float*    jk      = ws + 3 * NH;              // [N, 64]
  float*    alpha_s = ws + 4 * NH;              // [N]
  float*    alpha_d = alpha_s + GAT_N;          // [N]
  unsigned* mmax    = (unsigned*)(alpha_d + GAT_N); // [N]
  float*    denom   = (float*)(mmax + GAT_N);   // [N]
  float*    pbuf    = denom + GAT_N;            // [E + N]

  const int nodeWaveBlocks = (GAT_N * 32 + 255) / 256;             // wave per node
  const int edgeThrBlocks  = (GAT_ET + 255) / 256;                 // thread per edge
  const int edgeWaveBlocks = (int)(((size_t)GAT_ET * 32 + 255) / 256); // wave per edge
  const int nhBlocks       = (int)((NH + 255) / 256);

  for (int l = 0; l < 3; ++l) {
    const float* Ain = (l == 0) ? x : xin;
    const int    K   = (l == 0) ? GAT_FIN : GAT_H;
    const float* W   = (l == 0) ? W1 : (W23 + (size_t)(l - 1) * GAT_H * GAT_H);

    // One wave per 16-row strip, all 64 output cols (4 WMMA tiles) per wave.
    wmma_gemm_kernel<4><<<GAT_N / 16, 32, 0, stream>>>(
        Ain, W, hproj, K, GAT_H, GAT_H, nullptr);

    node_prep_kernel<<<nodeWaveBlocks, 256, 0, stream>>>(
        hproj, a_src + l * GAT_H, a_dst + l * GAT_H,
        alpha_s, alpha_d, mmax, denom, accum);

    edge_max_kernel<<<edgeThrBlocks, 256, 0, stream>>>(ei, alpha_s, alpha_d, mmax);
    edge_exp_kernel<<<edgeThrBlocks, 256, 0, stream>>>(ei, alpha_s, alpha_d, mmax, pbuf, denom);
    edge_agg_kernel<<<edgeWaveBlocks, 256, 0, stream>>>(ei, pbuf, denom, hproj, accum);

    act_jk_kernel<<<nhBlocks, 256, 0, stream>>>(accum, b + l * GAT_H, xin, jk, l == 0);
  }

  // Final projection: [N,64] @ [64,40] + bout, cols padded to 48 (3 tiles).
  wmma_gemm_kernel<3><<<GAT_N / 16, 32, 0, stream>>>(
      jk, Wout, out, GAT_H, GAT_OUT, GAT_OUT, bout);
}
